// InvariantPointAttention_65292092834419
// MI455X (gfx1250) — matrix operand order repaired and checked by reading, not compiled, not run on hardware
//
#include <hip/hip_runtime.h>
#include <hip/hip_bf16.h>

// ---------------------------------------------------------------------------
// InvariantPointAttention for MI455X (gfx1250).
// bf16 WMMA everywhere, fp32 accumulation, flash-fused attention with
// 32-keys-per-step (full K for every WMMA), b128 vector loads via bf16
// pre-converted / pre-transposed operand layouts.  Softmax reductions are
// done in-lane on the A-layout side of the LDS score bounce: one
// shfl_xor(16) per reduction instead of 4-stage ds_bpermute butterflies,
// and per-row state is 2 scalars (no spills).
// ---------------------------------------------------------------------------

typedef __attribute__((ext_vector_type(16))) __bf16 v16bf;
typedef __attribute__((ext_vector_type(8)))  __bf16 bf16x8;
typedef __attribute__((ext_vector_type(8)))  float  v8f;

#define DIM   256
#define NH    8
#define HDIM  32
#define NP    16
#define NB    2
#define SEQ   2048
#define MTOT  (NB * SEQ)          // 4096 rows
#define HP    (NH * NP)           // 128
#define QTPB  4                   // q-tiles (waves) per block in attention

static __device__ __forceinline__ __bf16 bfc(float x) { return (__bf16)x; }

static __device__ __forceinline__ bf16x8 ld8(const __bf16* p) {
  return *(const bf16x8*)p;
}

static __device__ __forceinline__ v16bf cat16(bf16x8 a, bf16x8 b) {
  v16bf r;
#pragma unroll
  for (int i = 0; i < 8; ++i) { r[i] = a[i]; r[i + 8] = b[i]; }
  return r;
}

static __device__ __forceinline__ v16bf zero16() {
  v16bf r;
#pragma unroll
  for (int i = 0; i < 16; ++i) r[i] = (__bf16)0.0f;
  return r;
}

static __device__ __forceinline__ v8f wmma_bf16(v16bf a, v16bf b, v8f c) {
  // (neg_a, A, neg_b, B, c_mod, C, reuse_a, reuse_b)
  return __builtin_amdgcn_wmma_f32_16x16x32_bf16(false, a, false, b,
                                                 (short)0, c, false, false);
}

// ---------------------------------------------------------------------------
// fp32 -> bf16 elementwise
// ---------------------------------------------------------------------------
__global__ void cvt_bf16_kernel(const float* __restrict__ x,
                                __bf16* __restrict__ y, int n) {
  int i = blockIdx.x * blockDim.x + threadIdx.x;
  if (i < n) y[i] = bfc(x[i]);
}

// ---------------------------------------------------------------------------
// W (K x N fp32, row-major) -> WT (N x K bf16) so B-fragments are contiguous.
// ---------------------------------------------------------------------------
__global__ void transpose_bf16_kernel(const float* __restrict__ W,
                                      __bf16* __restrict__ WT, int K, int N) {
  int i = blockIdx.x * blockDim.x + threadIdx.x;
  if (i >= K * N) return;
  int kk = i / N;
  int n  = i - kk * N;
  WT[(size_t)n * K + kk] = bfc(W[i]);
}

// ---------------------------------------------------------------------------
// GEMM  Y = Xbf(M x K) @ WT(N x K)^T + bias.  One wave per 16x64 tile
// (A fragment reused across 4 WMMAs).  All loads are 16-byte b128.
// MODE 0: bf16 row-major out.  MODE 1: bf16 transposed out (N x MTOT).
// MODE 2: fp32 row-major out.
// ---------------------------------------------------------------------------
template <int MODE>
__global__ __launch_bounds__(32, 1)
void gemm_bf16_kernel(const __bf16* __restrict__ X,
                      const __bf16* __restrict__ WT,
                      const float* __restrict__ bias,
                      void* __restrict__ Yv,
                      int Nn, int Kn) {
  const int l  = threadIdx.x & 31;
  const int hi = l >> 4;
  const int lo = l & 15;
  const int m0 = blockIdx.x * 16;
  const int n0 = blockIdx.y * 64;

  v8f acc[4] = {{}, {}, {}, {}};
  for (int kb = 0; kb < Kn; kb += 32) {
    // A 16x32: lane=M, elements = two runs K=[hi*8, +8) and K=[hi*8+16, +8)
    const __bf16* xr = X + (size_t)(m0 + lo) * Kn + kb + hi * 8;
    v16bf a = cat16(ld8(xr), ld8(xr + 16));
#pragma unroll
    for (int t = 0; t < 4; ++t) {
      // B 32x16: lane=N, elements = two runs K=[hi*8, +8) and K=[hi*8+16, +8)
      const __bf16* wr = WT + (size_t)(n0 + t * 16 + lo) * Kn + kb + hi * 8;
      v16bf bm = cat16(ld8(wr), ld8(wr + 16));
      acc[t] = wmma_bf16(a, bm, acc[t]);
    }
  }

#pragma unroll
  for (int t = 0; t < 4; ++t) {
    const int n = n0 + t * 16 + lo;
    const float bv = bias[n];
    if (MODE == 1) {                    // rows for fixed col are consecutive
      bf16x8 o;
#pragma unroll
      for (int r = 0; r < 8; ++r) o[r] = bfc(acc[t][r] + bv);
      *(bf16x8*)((__bf16*)Yv + (size_t)n * MTOT + m0 + 8 * hi) = o;
    } else {
#pragma unroll
      for (int r = 0; r < 8; ++r) {
        int row = r + 8 * hi;
        size_t idx = (size_t)(m0 + row) * Nn + n;
        if (MODE == 0) ((__bf16*)Yv)[idx] = bfc(acc[t][r] + bv);
        else           ((float*)Yv)[idx]  = acc[t][r] + bv;
      }
    }
  }
}

// ---------------------------------------------------------------------------
// Point projections (K=3): rep = coords[..,1,:] -> pq (row-major bf16),
// pqT (HP x MTOT bf16, for B-operand use), pk (row-major bf16).
// ---------------------------------------------------------------------------
__global__ void point_proj_kernel(const float* __restrict__ coords,
                                  const float* __restrict__ Wpq,
                                  const float* __restrict__ bpq,
                                  const float* __restrict__ Wpk,
                                  const float* __restrict__ bpk,
                                  __bf16* __restrict__ pq,
                                  __bf16* __restrict__ pqT,
                                  __bf16* __restrict__ pk) {
  int idx = blockIdx.x * blockDim.x + threadIdx.x;
  if (idx >= MTOT * HP) return;
  int row = idx / HP;
  int col = idx - row * HP;
  const float* rep = coords + ((size_t)row * 3 + 1) * 3;   // middle row of 3x3
  float r0 = rep[0], r1 = rep[1], r2 = rep[2];
  float pqv = r0 * Wpq[0 * HP + col] + r1 * Wpq[1 * HP + col] +
              r2 * Wpq[2 * HP + col] + bpq[col];
  float pkv = r0 * Wpk[0 * HP + col] + r1 * Wpk[1 * HP + col] +
              r2 * Wpk[2 * HP + col] + bpk[col];
  pq [idx] = bfc(pqv);
  pqT[(size_t)col * MTOT + row] = bfc(pqv);
  pk [idx] = bfc(pkv);
}

// ---------------------------------------------------------------------------
// Fused flash attention, 32 keys per step.
//  - scores = qk/sqrt(32) + pq.pk*0.25 + bias[h]; col-mask applied in C layout
//  - raw fp32 scores bounce through per-wave LDS tile into A layout, where
//    each lane owns half a row as two 8-col runs: row mask, max/sum reductions
//    (in-lane + one shfl_xor(16)), exp, and bf16 P-fragment are all done
//    in-lane.  Per-row correction factors broadcast via a 16-float LDS line.
// Emits seq_ctx as bf16 (A-operand of the Wo GEMM) and pt_ctx as fp32.
// ---------------------------------------------------------------------------
__global__ __launch_bounds__(32 * QTPB, 1)
void ipa_attn_kernel(const __bf16* __restrict__ q,
                     const __bf16* __restrict__ k,
                     const __bf16* __restrict__ vT,
                     const __bf16* __restrict__ pq,
                     const __bf16* __restrict__ pqT,
                     const __bf16* __restrict__ pk,
                     const int*   __restrict__ mask,
                     const float* __restrict__ attn_bias,
                     __bf16* __restrict__ seq_ctx,
                     float*  __restrict__ pt_ctx) {
  __shared__ float sls[QTPB][16 * 32];   // per-wave fp32 score tile
  __shared__ float brd[QTPB][32];        // [0:16) corr, [16:32) row-sum bcast

  const int l    = threadIdx.x & 31;
  const int wave = threadIdx.x >> 5;
  const int hi   = l >> 4;
  const int lo   = l & 15;

  const int bh = blockIdx.x;
  const int b  = bh / NH;
  const int h  = bh - b * NH;
  const int i0 = (blockIdx.y * QTPB + wave) * 16;
  const int mb = b * SEQ;

  const float scale  = 0.17677669529663687f;  // 1/sqrt(32)
  const float bias_h = attn_bias[h];

  // ---- A fragments (live for whole key loop) ----
  const __bf16* qp = q + (size_t)(mb + i0 + lo) * DIM + h * HDIM + hi * 8;
  v16bf aq = cat16(ld8(qp), ld8(qp + 16));
  const __bf16* pp = pq + (size_t)(mb + i0 + lo) * HP + h * NP + hi * 8;
  bf16x8 z8;
#pragma unroll
  for (int i = 0; i < 8; ++i) z8[i] = (__bf16)0.0f;
  v16bf apq = cat16(ld8(pp), z8);           // K = hi*8+e valid (<16), rest 0

  const int maskiA = mask[mb + i0 + lo];    // row mask, A-layout (row = lo)

  v8f o0 = {}, o1 = {}, op = {};
  float mrowA = -1e30f;                     // running max / sum for row lo
  float lrowA = 0.0f;

  for (int j0 = 0; j0 < SEQ; j0 += 32) {
    // ---- scores for two 16-key tiles (keys j0.. and j0+16..) ----
    const __bf16* kpa = k + (size_t)(mb + j0 + lo) * DIM + h * HDIM + hi * 8;
    const __bf16* kpb = kpa + (size_t)16 * DIM;
    v16bf bka = cat16(ld8(kpa), ld8(kpa + 16));
    v16bf bkb = cat16(ld8(kpb), ld8(kpb + 16));
    v16bf bpka = zero16(), bpkb = zero16();
    {                                       // K = hi*8+e valid (<16), rest 0; N = key
      const __bf16* ka  = pk + (size_t)(mb + j0 + lo) * HP + h * NP + hi * 8;
      const __bf16* kbp = ka + (size_t)16 * HP;
      bpka = cat16(ld8(ka),  z8);
      bpkb = cat16(ld8(kbp), z8);
    }

    v8f zc = {};
    v8f s1a = wmma_bf16(aq,  bka,  zc);
    v8f s2a = wmma_bf16(apq, bpka, zc);
    v8f s1b = wmma_bf16(aq,  bkb,  zc);
    v8f s2b = wmma_bf16(apq, bpkb, zc);

    if (j0 + 32 < SEQ) {                    // prefetch next key tile
      __builtin_prefetch(k + (size_t)(mb + j0 + 32 + lo) * DIM + h * HDIM, 0, 1);
      __builtin_prefetch(vT + (size_t)(h * HDIM + lo) * MTOT + mb + j0 + 32, 0, 1);
    }

    const int mja = mask[mb + j0 + lo];
    const int mjb = mask[mb + j0 + 16 + lo];

    // ---- combine + col-mask in C layout, store fp32 scores to LDS ----
#pragma unroll
    for (int r = 0; r < 8; ++r) {
      float sa = s1a[r] * scale + s2a[r] * 0.25f + bias_h;
      float sb = s1b[r] * scale + s2b[r] * 0.25f + bias_h;
      sa = mja ? sa : -1e9f;
      sb = mjb ? sb : -1e9f;
      sls[wave][(r + 8 * hi) * 32 + lo]      = sa;
      sls[wave][(r + 8 * hi) * 32 + 16 + lo] = sb;
    }
    __syncthreads();

    // ---- A-layout: lane owns row lo, cols [hi*8,+8) and [hi*8+16,+8) ----
    float s16[16];
    {
      const float4* p0 = (const float4*)&sls[wave][lo * 32 + hi * 8];
      const float4* p1 = (const float4*)&sls[wave][lo * 32 + hi * 8 + 16];
      float4 c0 = p0[0], c1 = p0[1], c2 = p1[0], c3 = p1[1];
      s16[0]=c0.x; s16[1]=c0.y; s16[2]=c0.z; s16[3]=c0.w;
      s16[4]=c1.x; s16[5]=c1.y; s16[6]=c1.z; s16[7]=c1.w;
      s16[8]=c2.x; s16[9]=c2.y; s16[10]=c2.z; s16[11]=c2.w;
      s16[12]=c3.x; s16[13]=c3.y; s16[14]=c3.z; s16[15]=c3.w;
    }
    if (!maskiA) {
#pragma unroll
      for (int e = 0; e < 16; ++e) s16[e] = -1e9f;
    }

    float mx = s16[0];
#pragma unroll
    for (int e = 1; e < 16; ++e) mx = fmaxf(mx, s16[e]);
    mx = fmaxf(mx, __shfl_xor(mx, 16));     // combine the two half-rows
    float nm   = fmaxf(mrowA, mx);
    float corr = __expf(mrowA - nm);
    mrowA = nm;

    v16bf ap;
    float rs = 0.0f;
#pragma unroll
    for (int e = 0; e < 16; ++e) {
      float pe = __expf(s16[e] - nm);
      rs += pe;
      ap[e] = bfc(pe);                      // element order == A-layout K order
    }
    rs += __shfl_xor(rs, 16);
    lrowA = lrowA * corr + rs;

    // ---- broadcast per-row correction to C-layout accumulators ----
    if (l < 16) brd[wave][lo] = corr;
    __syncthreads();
    float c8[8];
    {
      float4 x0 = *(const float4*)&brd[wave][8 * hi];
      float4 x1 = *(const float4*)&brd[wave][8 * hi + 4];
      c8[0]=x0.x; c8[1]=x0.y; c8[2]=x0.z; c8[3]=x0.w;
      c8[4]=x1.x; c8[5]=x1.y; c8[6]=x1.z; c8[7]=x1.w;
    }
#pragma unroll
    for (int r = 0; r < 8; ++r) { o0[r] *= c8[r]; o1[r] *= c8[r]; op[r] *= c8[r]; }

    // ---- P @ V and P @ pq over full K=32 keys (transposed operands) ----
    const __bf16* v0p = vT + (size_t)(h * HDIM + lo) * MTOT + mb + j0 + hi * 8;
    const __bf16* v1p = v0p + (size_t)16 * MTOT;
    const __bf16* pqp = pqT + (size_t)(h * NP + lo) * MTOT + mb + j0 + hi * 8;
    v16bf bv0  = cat16(ld8(v0p), ld8(v0p + 16));
    v16bf bv1  = cat16(ld8(v1p), ld8(v1p + 16));
    v16bf bpqf = cat16(ld8(pqp), ld8(pqp + 16));
    o0 = wmma_bf16(ap, bv0,  o0);
    o1 = wmma_bf16(ap, bv1,  o1);
    op = wmma_bf16(ap, bpqf, op);
  }

  // ---- broadcast row sums, normalize, write contexts ----
  if (l < 16) brd[wave][16 + lo] = lrowA;
  __syncthreads();
  float l8[8];
  {
    float4 x0 = *(const float4*)&brd[wave][16 + 8 * hi];
    float4 x1 = *(const float4*)&brd[wave][16 + 8 * hi + 4];
    l8[0]=x0.x; l8[1]=x0.y; l8[2]=x0.z; l8[3]=x0.w;
    l8[4]=x1.x; l8[5]=x1.y; l8[6]=x1.z; l8[7]=x1.w;
  }
#pragma unroll
  for (int r = 0; r < 8; ++r) {
    int row = i0 + r + 8 * hi;
    float inv = 1.0f / l8[r];
    seq_ctx[(size_t)(mb + row) * DIM + h * HDIM + lo]      = bfc(o0[r] * inv);
    seq_ctx[(size_t)(mb + row) * DIM + h * HDIM + 16 + lo] = bfc(o1[r] * inv);
    pt_ctx [(size_t)(mb + row) * HP  + h * NP   + lo]      = op[r] * inv;
  }
}

// ---------------------------------------------------------------------------
// point_out = pt_ctx(MTOT x 128) @ Wpo(128 x 3) + bpo   (N=3: scalar VALU)
// ---------------------------------------------------------------------------
__global__ void point_out_kernel(const float* __restrict__ pt_ctx,
                                 const float* __restrict__ Wpo,
                                 const float* __restrict__ bpo,
                                 float* __restrict__ out) {
  int idx = blockIdx.x * blockDim.x + threadIdx.x;
  if (idx >= MTOT * 3) return;
  int row = idx / 3;
  int c   = idx - row * 3;
  float acc = bpo[c];
  const float* x = pt_ctx + (size_t)row * HP;
#pragma unroll 8
  for (int p = 0; p < HP; ++p) acc += x[p] * Wpo[p * 3 + c];
  out[idx] = acc;
}

// ---------------------------------------------------------------------------
extern "C" void kernel_launch(void* const* d_in, const int* in_sizes, int n_in,
                              void* d_out, int out_size, void* d_ws,
                              size_t ws_size, hipStream_t stream) {
  (void)in_sizes; (void)n_in; (void)out_size; (void)ws_size;

  const float* seq    = (const float*)d_in[0];
  const float* coords = (const float*)d_in[1];
  /* d_in[2] frames: unused by reference */
  const int*   mask   = (const int*)  d_in[3];
  const float* Wq  = (const float*)d_in[4];  const float* bq  = (const float*)d_in[5];
  const float* Wk  = (const float*)d_in[6];  const float* bk  = (const float*)d_in[7];
  const float* Wv  = (const float*)d_in[8];  const float* bv  = (const float*)d_in[9];
  const float* Wpq = (const float*)d_in[10]; const float* bpq = (const float*)d_in[11];
  const float* Wpk = (const float*)d_in[12]; const float* bpk = (const float*)d_in[13];
  const float* Wo  = (const float*)d_in[14]; const float* bo  = (const float*)d_in[15];
  const float* Wpo = (const float*)d_in[16]; const float* bpo = (const float*)d_in[17];
  const float* attn_bias = (const float*)d_in[18];

  // ---- workspace layout (16B aligned slices) ----
  char* wsb = (char*)d_ws;
  size_t off = 0;
  auto take = [&](size_t bytes) {
    char* p = wsb + off;
    off += (bytes + 15) & ~(size_t)15;
    return p;
  };
  __bf16* xbf   = (__bf16*)take((size_t)MTOT * DIM * 2);
  __bf16* wqT   = (__bf16*)take((size_t)DIM * DIM * 2);
  __bf16* wkT   = (__bf16*)take((size_t)DIM * DIM * 2);
  __bf16* wvT   = (__bf16*)take((size_t)DIM * DIM * 2);
  __bf16* woT   = (__bf16*)take((size_t)DIM * DIM * 2);
  __bf16* qbf   = (__bf16*)take((size_t)MTOT * DIM * 2);
  __bf16* kbf   = (__bf16*)take((size_t)MTOT * DIM * 2);
  __bf16* vTbf  = (__bf16*)take((size_t)DIM * MTOT * 2);
  __bf16* pqbf  = (__bf16*)take((size_t)MTOT * HP * 2);
  __bf16* pqTbf = (__bf16*)take((size_t)HP * MTOT * 2);
  __bf16* pkbf  = (__bf16*)take((size_t)MTOT * HP * 2);
  __bf16* sctx  = (__bf16*)take((size_t)MTOT * DIM * 2);
  float*  pctx  = (float*) take((size_t)MTOT * HP * 4);

  // ---- bf16 conversion / weight transposition ----
  int nx = MTOT * DIM;
  cvt_bf16_kernel<<<(nx + 255) / 256, 256, 0, stream>>>(seq, xbf, nx);
  int nw = DIM * DIM;
  transpose_bf16_kernel<<<(nw + 255) / 256, 256, 0, stream>>>(Wq, wqT, DIM, DIM);
  transpose_bf16_kernel<<<(nw + 255) / 256, 256, 0, stream>>>(Wk, wkT, DIM, DIM);
  transpose_bf16_kernel<<<(nw + 255) / 256, 256, 0, stream>>>(Wv, wvT, DIM, DIM);
  transpose_bf16_kernel<<<(nw + 255) / 256, 256, 0, stream>>>(Wo, woT, DIM, DIM);

  // ---- projections (WMMA GEMMs; v written transposed) ----
  dim3 gg(MTOT / 16, DIM / 64);
  gemm_bf16_kernel<0><<<gg, 32, 0, stream>>>(xbf, wqT, bq, qbf,  DIM, DIM);
  gemm_bf16_kernel<0><<<gg, 32, 0, stream>>>(xbf, wkT, bk, kbf,  DIM, DIM);
  gemm_bf16_kernel<1><<<gg, 32, 0, stream>>>(xbf, wvT, bv, vTbf, DIM, DIM);

  int np = MTOT * HP;
  point_proj_kernel<<<(np + 255) / 256, 256, 0, stream>>>(
      coords, Wpq, bpq, Wpk, bpk, pqbf, pqTbf, pkbf);

  // ---- fused attention ----
  dim3 ga(NB * NH, SEQ / (16 * QTPB));
  ipa_attn_kernel<<<ga, 32 * QTPB, 0, stream>>>(
      qbf, kbf, vTbf, pqbf, pqTbf, pkbf, mask, attn_bias, sctx, pctx);

  // ---- output projections ----
  gemm_bf16_kernel<2><<<gg, 32, 0, stream>>>(sctx, woT, bo, (float*)d_out,
                                             DIM, DIM);
  int npo = MTOT * 3;
  point_out_kernel<<<(npo + 255) / 256, 256, 0, stream>>>(
      pctx, Wpo, bpo, (float*)d_out + (size_t)MTOT * DIM);
}
